// SharedTTLinearLayer_17463337025854
// MI455X (gfx1250) — matrix operand order repaired and checked by reading, not compiled, not run on hardware
//
#include <hip/hip_runtime.h>
#include <hip/hip_bf16.h>

typedef __attribute__((ext_vector_type(16))) _Float16 v16h;
typedef __attribute__((ext_vector_type(8)))  _Float16 v8h;
typedef __attribute__((ext_vector_type(8)))  float    v8f;

#define B_   4096
#define T_   8
#define IN_  1024
#define OUT_ 1024
#define R_   32

#define BM 128
#define BN 128
#define BK 32
#define LDST 40   // LDS row stride in halves (80B, conflict-avoiding pad)

// ---- CDNA5 async copy helpers (memory -> LDS, tracked by ASYNCcnt) --------
__device__ __forceinline__ void async_load_b128_to_lds(unsigned lds_byte_addr,
                                                       const void* gaddr) {
  // GLOBAL_LOAD_ASYNC_TO_LDS_B128: VDST = LDS byte address, VADDR = 64-bit
  // global address (GV mode, no SADDR).
  asm volatile("global_load_async_to_lds_b128 %0, %1, off"
               :: "v"(lds_byte_addr), "v"(gaddr)
               : "memory");
}

__device__ __forceinline__ void wait_asynccnt0() {
#if __has_builtin(__builtin_amdgcn_s_wait_asynccnt)
  __builtin_amdgcn_s_wait_asynccnt(0);
#else
  asm volatile("s_wait_asynccnt 0x0" ::: "memory");
#endif
}

// ---------------------------------------------------------------------------
// Stage 1: w1[t,i,k] = sum_j first[t,j] * middle[j,i,k]   (fp32, tiny)
// ---------------------------------------------------------------------------
__global__ __launch_bounds__(256) void tt_stage1(
    const float* __restrict__ first,
    const float* __restrict__ middle,
    float* __restrict__ w1) {
  const int idx = blockIdx.x * blockDim.x + threadIdx.x;  // t*IN + i
  const int t = idx >> 10;
  const int i = idx & (IN_ - 1);
  float acc[R_];
#pragma unroll
  for (int k = 0; k < R_; ++k) acc[k] = 0.f;
  for (int j = 0; j < R_; ++j) {
    const float fj = first[t * R_ + j];
    const float* mrow = middle + ((size_t)j * IN_ + i) * R_;
#pragma unroll
    for (int k = 0; k < R_; ++k) acc[k] += fj * mrow[k];
  }
  float* o = w1 + (size_t)idx * R_;
#pragma unroll
  for (int k = 0; k < R_; ++k) o[k] = acc[k];
}

// ---------------------------------------------------------------------------
// Stage 2: wh[t][o][i] = f16( sum_k w1[t,i,k] * task[k,o] )   (o-major)
// ---------------------------------------------------------------------------
__global__ __launch_bounds__(256) void tt_stage2(
    const float* __restrict__ w1,
    const float* __restrict__ task,
    _Float16* __restrict__ wh) {
  const int o = blockIdx.x;
  const int t = blockIdx.y;
  float tc[R_];
#pragma unroll
  for (int k = 0; k < R_; ++k) tc[k] = task[k * OUT_ + o];
  _Float16* dst = wh + ((size_t)t * OUT_ + o) * IN_;
  const float* src = w1 + (size_t)t * IN_ * R_;
  for (int i = threadIdx.x; i < IN_; i += blockDim.x) {
    const float* wr = src + (size_t)i * R_;
    float acc = 0.f;
#pragma unroll
    for (int k = 0; k < R_; ++k) acc += wr[k] * tc[k];
    dst[i] = (_Float16)acc;
  }
}

// ---------------------------------------------------------------------------
// Main GEMM per task t: C[b,o] = sum_i X[b,i] * W[i,o]
// 256 thr = 8 waves, block tile 128x128x32, wave tile 32x64 (2x4 frags).
// Double-buffered LDS; W tile via GLOBAL_LOAD_ASYNC_TO_LDS_B128 (ASYNCcnt),
// X tile via f32 loads + v_cvt + ds_store_b128. One barrier per K-step.
// ---------------------------------------------------------------------------
__global__ __launch_bounds__(256) void tt_gemm(
    const float* __restrict__ x,
    const _Float16* __restrict__ wh,
    float* __restrict__ out) {
  __shared__ __align__(16) _Float16 sX[2][BM * LDST];
  __shared__ __align__(16) _Float16 sW[2][BN * LDST];

  const int tid = threadIdx.x;
  const int lane16 = tid & 15;
  const int halfsel = (tid >> 4) & 1;   // lane/16 within the wave
  const int wave = tid >> 5;
  const int wm = wave & 3;              // 4 waves along M (32 rows each)
  const int wn = wave >> 2;             // 2 waves along N (64 cols each)

  const int t  = blockIdx.z;
  const int b0 = blockIdx.y * BM;
  const int n0 = blockIdx.x * BN;

  const v8f zero = {0.f, 0.f, 0.f, 0.f, 0.f, 0.f, 0.f, 0.f};
  v8f c[2][4];
#pragma unroll
  for (int ms = 0; ms < 2; ++ms)
#pragma unroll
    for (int ns = 0; ns < 4; ++ns) c[ms][ns] = zero;

  // staging coords: 2 threads per tile row, 16 elements each
  const int xrow = tid >> 1, xcg = tid & 1;
  const int stageOff = xrow * LDST + xcg * 16;           // element offset
  const float*    xbase = x  + (((size_t)(b0 + xrow)) * T_ + t) * IN_ + xcg * 16;
  const _Float16* wbase = wh + (((size_t)t * OUT_) + (n0 + xrow)) * IN_ + xcg * 16;

  // ---- prologue: stage K-step 0 into buffer 0 ----
  {
    async_load_b128_to_lds((unsigned)(uintptr_t)&sW[0][stageOff],     wbase);
    async_load_b128_to_lds((unsigned)(uintptr_t)&sW[0][stageOff + 8], wbase + 8);
    const float4* p = (const float4*)xbase;
    float f[16];
#pragma unroll
    for (int q = 0; q < 4; ++q) {
      float4 v = p[q];
      f[4*q+0] = v.x; f[4*q+1] = v.y; f[4*q+2] = v.z; f[4*q+3] = v.w;
    }
    v8h h0, h1;
#pragma unroll
    for (int e = 0; e < 8; ++e) { h0[e] = (_Float16)f[e]; h1[e] = (_Float16)f[8+e]; }
    *(v8h*)&sX[0][stageOff]     = h0;
    *(v8h*)&sX[0][stageOff + 8] = h1;
    wait_asynccnt0();
  }
  __syncthreads();

  const int KITER = IN_ / BK;  // 32
  for (int it = 0; it < KITER; ++it) {
    const int cur = it & 1;
    const int nxt = cur ^ 1;
    const bool havenext = (it + 1) < KITER;
    const int kkn = (it + 1) * BK;

    // ---- issue next tile's fetches (overlap with WMMAs) ----
    float4 xf0, xf1, xf2, xf3;
    if (havenext) {
      async_load_b128_to_lds((unsigned)(uintptr_t)&sW[nxt][stageOff],     wbase + kkn);
      async_load_b128_to_lds((unsigned)(uintptr_t)&sW[nxt][stageOff + 8], wbase + kkn + 8);
      const float4* p = (const float4*)(xbase + kkn);
      xf0 = p[0]; xf1 = p[1]; xf2 = p[2]; xf3 = p[3];
    }

    // ---- compute on current buffer ----
    v16h a[2], b[4];
#pragma unroll
    for (int ms = 0; ms < 2; ++ms) {
      const int row = wm * 32 + ms * 16 + lane16;                  // M = lane%16
      v8h lo = *(const v8h*)&sX[cur][row * LDST + halfsel * 8];        // K=h*8..+7
      v8h hi = *(const v8h*)&sX[cur][row * LDST + 16 + halfsel * 8];   // K=16+h*8..+7
#pragma unroll
      for (int e = 0; e < 8; ++e) { a[ms][e] = lo[e]; a[ms][8+e] = hi[e]; }
    }
#pragma unroll
    for (int ns = 0; ns < 4; ++ns) {
      const int col = wn * 64 + ns * 16 + lane16;                  // N = lane%16
      v8h lo = *(const v8h*)&sW[cur][col * LDST + halfsel * 8];        // K=h*8..+7
      v8h hi = *(const v8h*)&sW[cur][col * LDST + 16 + halfsel * 8];   // K=16+h*8..+7
#pragma unroll
      for (int e = 0; e < 8; ++e) { b[ns][e] = lo[e]; b[ns][8+e] = hi[e]; }
    }
#pragma unroll
    for (int ms = 0; ms < 2; ++ms)
#pragma unroll
      for (int ns = 0; ns < 4; ++ns)
        c[ms][ns] = __builtin_amdgcn_wmma_f32_16x16x32_f16(
            false, a[ms], false, b[ns], (short)0, c[ms][ns], false, false);

    // ---- finish staging next buffer ----
    if (havenext) {
      float f[16];
      f[0]=xf0.x; f[1]=xf0.y; f[2]=xf0.z; f[3]=xf0.w;
      f[4]=xf1.x; f[5]=xf1.y; f[6]=xf1.z; f[7]=xf1.w;
      f[8]=xf2.x; f[9]=xf2.y; f[10]=xf2.z; f[11]=xf2.w;
      f[12]=xf3.x; f[13]=xf3.y; f[14]=xf3.z; f[15]=xf3.w;
      v8h h0, h1;
#pragma unroll
      for (int e = 0; e < 8; ++e) { h0[e] = (_Float16)f[e]; h1[e] = (_Float16)f[8+e]; }
      *(v8h*)&sX[nxt][stageOff]     = h0;
      *(v8h*)&sX[nxt][stageOff + 8] = h1;
      wait_asynccnt0();
    }
    __syncthreads();
  }

  // ---- epilogue: C VGPR r <-> M = half*8 + r, N = lane%16 ----
#pragma unroll
  for (int ms = 0; ms < 2; ++ms) {
#pragma unroll
    for (int ns = 0; ns < 4; ++ns) {
      const int N = n0 + wn * 64 + ns * 16 + lane16;
#pragma unroll
      for (int r = 0; r < 8; ++r) {
        const int M = b0 + wm * 32 + ms * 16 + halfsel * 8 + r;
        out[(((size_t)M) * T_ + t) * OUT_ + N] = c[ms][ns][r];
      }
    }
  }
}

// ---------------------------------------------------------------------------
extern "C" void kernel_launch(void* const* d_in, const int* in_sizes, int n_in,
                              void* d_out, int out_size, void* d_ws, size_t ws_size,
                              hipStream_t stream) {
  const float* x      = (const float*)d_in[0];  // [B,T,IN]
  const float* first  = (const float*)d_in[1];  // [T,R]
  const float* middle = (const float*)d_in[2];  // [R,IN,R]
  const float* task   = (const float*)d_in[3];  // [R,OUT]
  float* out = (float*)d_out;                   // [B,T,OUT]

  // workspace: wh f16 [T][OUT][IN] = 16MB, then w1 f32 [T][IN][R] = 1MB
  _Float16* wh = (_Float16*)d_ws;
  float* w1 = (float*)((char*)d_ws + (size_t)T_ * OUT_ * IN_ * sizeof(_Float16));

  tt_stage1<<<dim3((T_ * IN_) / 256), 256, 0, stream>>>(first, middle, w1);
  tt_stage2<<<dim3(OUT_, T_), 256, 0, stream>>>(w1, task, wh);
  tt_gemm<<<dim3(OUT_ / BN, B_ / BM, T_), 256, 0, stream>>>(x, wh, out);
}
